// AnimaHierarchical_52974126629426
// MI455X (gfx1250) — matrix operand order repaired and checked by reading, not compiled, not run on hardware
//
#include <hip/hip_runtime.h>

typedef __bf16 bf16;
typedef __attribute__((ext_vector_type(16))) __bf16 bf16x16;
typedef __attribute__((ext_vector_type(8)))  __bf16 bf16x8;
typedef __attribute__((ext_vector_type(8)))  float  f32x8;
typedef __attribute__((ext_vector_type(4)))  unsigned int u32x4;
typedef __attribute__((ext_vector_type(8)))  int i32x8;
typedef __attribute__((ext_vector_type(4)))  int i32x4;

#define T_SEQ   256
#define N_BATCH 32
#define DMODEL  512
#define DSTATE  16
#define DMETA   64
#define SENS    128
#define NBOT    256
#define NOUT    64
#define NCELL   19456          /* 6*512 + 2*8192 : z,r,alpha,beta,delta,gamma,W_B,W_C */
#define SZ      (N_BATCH*DMODEL)   /* 16384 */

#if __has_builtin(__builtin_amdgcn_tensor_load_to_lds)
#define ANIMA_TDM 1
#else
#define ANIMA_TDM 0
#endif

/* ---------------- small helpers ---------------- */

__device__ __forceinline__ bf16 f2bf(float f) {
  unsigned u = __builtin_bit_cast(unsigned int, f);
  unsigned r = u + 0x7FFFu + ((u >> 16) & 1u);
  unsigned short h = (unsigned short)(r >> 16);
  return __builtin_bit_cast(bf16, h);
}
__device__ __forceinline__ float sigf(float v) { return 1.f / (1.f + __expf(-v)); }
__device__ __forceinline__ float actf(float v, int act) {
  if (act == 1) return tanhf(v);
  if (act == 2) return sigf(v);
  return v;
}

/* A-fragment: 16-bit A 16x32, lanes 0-15 rows M, K pairs 0..7 & 16..23 (hi lanes +8) */
__device__ __forceinline__ bf16x16 load_a_frag(const bf16* A, int lda, int mt, int kt) {
  int lane = threadIdx.x & 31;
  const bf16* base = A + (size_t)(mt * 16 + (lane & 15)) * lda + kt * 32 + ((lane & 16) >> 1);
  bf16x8 lo = *(const bf16x8*)base;
  bf16x8 hi = *(const bf16x8*)(base + 16);
  bf16x16 v;
#pragma unroll
  for (int i = 0; i < 8; ++i) { v[i] = lo[i]; v[8 + i] = hi[i]; }
  return v;
}
/* B-fragment: 16-bit B 32x16, lane n = lane&15, K pairs 0..7 & 16..23 (hi lanes +8) */
__device__ __forceinline__ bf16x16 load_b_frag(const bf16* W, int ldw, int nt, int kt) {
  int lane = threadIdx.x & 31;
  const bf16* base = W + (size_t)(nt * 16 + (lane & 15)) * ldw + kt * 32 + ((lane & 16) >> 1);
  bf16x8 lo = *(const bf16x8*)base;
  bf16x8 hi = *(const bf16x8*)(base + 16);
  bf16x16 v;
#pragma unroll
  for (int i = 0; i < 8; ++i) { v[i] = lo[i]; v[8 + i] = hi[i]; }
  return v;
}

/* ---- Tensor Data Mover: one 2D descriptor copies a contiguous [32, l] bf16
   matrix from global into an LDS segment block (data_size = 8B units). ---- */
#if ANIMA_TDM
__device__ __forceinline__ void tdm_stage(const void* src, unsigned lds_off, int row_bf16) {
  unsigned ru = (unsigned)row_bf16 >> 2;              /* 8-byte units per row */
  unsigned long long ga = (unsigned long long)src;
  u32x4 g0;
  g0[0] = 1u;                                         /* count=1, user mode */
  g0[1] = lds_off;                                    /* D#.lds_addr */
  g0[2] = (unsigned)ga;                               /* global_addr[31:0] */
  g0[3] = (unsigned)((ga >> 32) & 0x1FFFFFFull) | (2u << 30); /* addr[56:32] | type=2 */
  i32x8 g1;
  g1[0] = (3 << 16);                                  /* data_size = 8B */
  g1[1] = (int)((ru & 0xFFFFu) << 16);                /* tensor_dim0[15:0] */
  g1[2] = (int)(((ru >> 16) & 0xFFFFu) | (32u << 16));/* dim0 hi | tensor_dim1=32 */
  g1[3] = (int)((ru & 0xFFFFu) << 16);                /* tile_dim0 = ru */
  g1[4] = 32;                                         /* tile_dim1 = 32 rows */
  g1[5] = (int)ru;                                    /* tensor_dim0_stride = ru */
  g1[6] = 0;
  g1[7] = 0;
  i32x4 z4 = (i32x4){0, 0, 0, 0};
#if defined(__clang_major__) && (__clang_major__ >= 23)
  i32x8 z8 = (i32x8){0, 0, 0, 0, 0, 0, 0, 0};
  __builtin_amdgcn_tensor_load_to_lds(g0, g1, z4, z4, z8, 0);
#else
  __builtin_amdgcn_tensor_load_to_lds(g0, g1, z4, z4, 0);
#endif
}
#endif

/* ---------------- parameter block ---------------- */

struct KP {
  const bf16 *wenc, *wfrom, *wgate, *ifrom, *cellA, *fasth, *metaw, *horw,
             *wabs, *wexp, *afrom, *agate, *wphi;
  const float *wenc_b, *wgate_b, *wng, *wnb, *abs_b, *exp_b, *ang, *anb,
              *agate_b, *phi_b, *fasth_b, *meta_b, *hor_b, *Dvec, *cng, *cnb;
  const float *cell_bias, *negA;
  const bf16 *xT_bf;
  float *xenc_f; bf16 *xenc_bf;
  float *Wst, *Ifst, *Ast, *Hmeta, *Hslow;
  bf16  *Wb, *Ifb, *Ab, *Hmb;
  float *wcoup_f, *wg_f, *iin_f, *cellraw, *hcand_f, *tau_f, *hso_f,
        *ain_f, *ag_f, *expand_f, *iout_f;
  bf16  *wcoup_bf, *iin_bf, *rh_bf, *iout_bf, *ain_bf, *abstr_bf;
  float *outp;
  unsigned *sync;
  float *stateF; int stateF_n;
  bf16  *stateB; int stateB_n;
};

struct SRC {
  const float *x, *wenc_w, *wfw, *wfi, *wfa, *wgate_w,
              *ifw, *ifi, *ifa, *abs_w, *exp_w, *afw, *afi, *afa,
              *agate_w, *phi_w,
              *fz_w, *fr_w, *fh_w, *wa_w, *wb_w, *wd_w, *Alog,
              *WB_w, *WB_b, *WC_w, *WC_b, *meta_w, *hor_w, *gam_w,
              *fz_b, *fr_b, *wa_b, *wb_b, *wd_b, *gam_b;
};

/* ---------------- grid barrier ---------------- */

__device__ void gsync(unsigned* sync) {
  __threadfence();
  __syncthreads();
  if (threadIdx.x == 0) {
    volatile unsigned* gen = sync + 1;
    unsigned g = *gen;
    if (atomicAdd(sync, 1u) == gridDim.x - 1) {
      sync[0] = 0;
      __threadfence();
      atomicAdd((unsigned*)(sync + 1), 1u);
    } else {
      while (*gen == g) { __builtin_amdgcn_s_sleep(1); }
    }
  }
  __syncthreads();
}

/* ------- generic 32xN GEMM: C = act(cat(a0,a1,a2) @ W^T + bias + addC) -------
   LDS layout is SEGMENT-MAJOR: block i holds the contiguous [32, l_i] matrix.
   All segment lengths are multiples of 32, so a K-tile never straddles blocks. */

__device__ void gemm3(bf16* sA,
                      const bf16* a0, int l0, const bf16* a1, int l1,
                      const bf16* a2, int l2,
                      const bf16* W, const float* bias, const float* addC,
                      float* Cf, int ldc, bf16* Cbf, int ldcb, int N, int act)
{
  const int K = l0 + l1 + l2;

#if ANIMA_TDM
  if (threadIdx.x < 32) {                 /* wave 0 issues the DMA descriptors */
    unsigned sbase = (unsigned)(unsigned long long)(const void*)sA;
    tdm_stage(a0, sbase, l0);
    if (l1) tdm_stage(a1, sbase + (unsigned)(N_BATCH * l0 * 2), l1);
    if (l2) tdm_stage(a2, sbase + (unsigned)(N_BATCH * (l0 + l1) * 2), l2);
#if __has_builtin(__builtin_amdgcn_s_wait_tensorcnt)
    __builtin_amdgcn_s_wait_tensorcnt(0);
#endif
  }
  __syncthreads();
#else
  {
    int t0 = N_BATCH * l0, t1 = N_BATCH * l1, t2 = N_BATCH * l2;
    for (int i = threadIdx.x * 8; i < t0; i += blockDim.x * 8)
      *(bf16x8*)(sA + i) = *(const bf16x8*)(a0 + i);
    for (int i = threadIdx.x * 8; i < t1; i += blockDim.x * 8)
      *(bf16x8*)(sA + t0 + i) = *(const bf16x8*)(a1 + i);
    for (int i = threadIdx.x * 8; i < t2; i += blockDim.x * 8)
      *(bf16x8*)(sA + t0 + t1 + i) = *(const bf16x8*)(a2 + i);
    __syncthreads();
  }
#endif

  const int kt0 = l0 >> 5, kt1 = (l0 + l1) >> 5;
  const int tt  = (N >> 5) << 1;          /* 2 M-tiles x (N/32) tile-pairs */
  const int gw  = (blockIdx.x * blockDim.x + threadIdx.x) >> 5;
  const int gws = (gridDim.x * blockDim.x) >> 5;
  const int ktn = K >> 5;

  for (int ti = gw; ti < tt; ti += gws) {
    int mt = ti & 1, np = ti >> 1;
    f32x8 acc0 = {}, acc1 = {};
    for (int kt = 0; kt < ktn; ++kt) {
      if (kt + 1 < ktn)
        __builtin_prefetch(W + (size_t)(np * 32) * K + (kt + 1) * 32, 0, 1);
      const bf16* ab; int lda, klo;
      if (kt < kt0)      { ab = sA;                          lda = l0; klo = kt; }
      else if (kt < kt1) { ab = sA + N_BATCH * l0;           lda = l1; klo = kt - kt0; }
      else               { ab = sA + N_BATCH * (l0 + l1);    lda = l2; klo = kt - kt1; }
      bf16x16 a  = load_a_frag(ab, lda, mt, klo);
      bf16x16 b0 = load_b_frag(W, K, np * 2, kt);
      bf16x16 b1 = load_b_frag(W, K, np * 2 + 1, kt);
      acc0 = __builtin_amdgcn_wmma_f32_16x16x32_bf16(false, a, false, b0,
                                                     (short)0, acc0, false, false);
      acc1 = __builtin_amdgcn_wmma_f32_16x16x32_bf16(false, a, false, b1,
                                                     (short)0, acc1, false, false);
    }
    int lane = threadIdx.x & 31;
    int mb = (mt << 4) + ((lane & 16) >> 1);
#pragma unroll
    for (int q = 0; q < 2; ++q) {
      f32x8 acc = q ? acc1 : acc0;
      int n = ((np * 2 + q) << 4) + (lane & 15);
      float bv = bias ? bias[n] : 0.f;
#pragma unroll
      for (int r = 0; r < 8; ++r) {
        int m = mb + r;
        float v = acc[r] + bv;
        if (addC) v += addC[m * N + n];
        v = actf(v, act);
        if (Cf)  Cf[(size_t)m * ldc + n] = v;
        if (Cbf) Cbf[(size_t)m * ldcb + n] = f2bf(v);
      }
    }
  }
}

/* ---------------- element-wise phases ---------------- */

__device__ void rowln_gate(const float* g, const float* a, const float* bold,
                           const float* ng, const float* nb,
                           float* outf, bf16* outb)
{
  int gw  = (blockIdx.x * blockDim.x + threadIdx.x) >> 5;
  int gws = (gridDim.x * blockDim.x) >> 5;
  int lane = threadIdx.x & 31;
  for (int row = gw; row < N_BATCH; row += gws) {
    float vals[16];
    float s = 0.f;
#pragma unroll
    for (int i = 0; i < 16; ++i) {
      int d = lane + i * 32, idx = row * DMODEL + d;
      float gg = g[idx];
      float v = gg * a[idx] + (1.f - gg) * bold[idx];
      vals[i] = v; s += v;
    }
    for (int off = 16; off >= 1; off >>= 1) s += __shfl_xor(s, off, 32);
    float mean = s * (1.f / DMODEL);
    float vs = 0.f;
#pragma unroll
    for (int i = 0; i < 16; ++i) { float d0 = vals[i] - mean; vs += d0 * d0; }
    for (int off = 16; off >= 1; off >>= 1) vs += __shfl_xor(vs, off, 32);
    float rs = rsqrtf(vs * (1.f / DMODEL) + 1e-5f);
#pragma unroll
    for (int i = 0; i < 16; ++i) {
      int d = lane + i * 32, idx = row * DMODEL + d;
      float y = (vals[i] - mean) * rs * ng[d] + nb[d];
      outf[idx] = y;
      if (outb) outb[idx] = f2bf(y);
    }
  }
}

__device__ void cell_slow(const float* cr, const float* iin, const float* hf_cur,
                          const float* hslow_cur, const float* negA,
                          float* hslow_nxt, float* hso, bf16* rh_bf)
{
  int tid = blockIdx.x * blockDim.x + threadIdx.x;
  int nth = gridDim.x * blockDim.x;
  for (int idx = tid; idx < SZ; idx += nth) {
    int b = idx >> 9, d = idx & 511;
    const float* c = cr + (size_t)b * NCELL;
    float r     = sigf(c[512 + d]);
    float alpha = sigf(c[1024 + d]);
    float beta  = sigf(c[1536 + d]);
    float delta = 0.1f * logf(1.f + __expf(c[2048 + d]));
    rh_bf[idx] = f2bf(r * hf_cur[idx]);
    float x = iin[idx];
    float acc = 0.f;
#pragma unroll
    for (int s = 0; s < DSTATE; ++s) {
      float Abar = __expf(delta * negA[d * DSTATE + s]);
      float Bm = c[3072 + d * DSTATE + s];
      float hn = alpha * Abar * hslow_cur[(size_t)idx * DSTATE + s]
               + beta * (delta * Bm) * x;
      hslow_nxt[(size_t)idx * DSTATE + s] = hn;
      acc += c[11264 + d * DSTATE + s] * hn;
    }
    hso[idx] = acc;
  }
}

__device__ void cell_mix(const float* cr, const float* hcand, const float* hf_cur,
                         const float* tau, const float* hso, const float* iin,
                         const float* Dv, const float* ng, const float* nb,
                         float* hf_nxt, bf16* hfb_nxt, float* iout, bf16* ioutb)
{
  int gw  = (blockIdx.x * blockDim.x + threadIdx.x) >> 5;
  int gws = (gridDim.x * blockDim.x) >> 5;
  int lane = threadIdx.x & 31;
  for (int row = gw; row < N_BATCH; row += gws) {
    float vals[16];
    float s = 0.f;
#pragma unroll
    for (int i = 0; i < 16; ++i) {
      int d = lane + i * 32, idx = row * DMODEL + d;
      float z = sigf(cr[(size_t)row * NCELL + d]);
      float hfn = (1.f - z) * hf_cur[idx] * 0.9f + z * hcand[idx];
      hf_nxt[idx] = hfn; hfb_nxt[idx] = f2bf(hfn);
      float tt = tau[idx];
      float v = (1.f - tt) * hfn + tt * hso[idx] + Dv[d] * iin[idx];
      vals[i] = v; s += v;
    }
    for (int off = 16; off >= 1; off >>= 1) s += __shfl_xor(s, off, 32);
    float mean = s * (1.f / DMODEL);
    float vs = 0.f;
#pragma unroll
    for (int i = 0; i < 16; ++i) { float d0 = vals[i] - mean; vs += d0 * d0; }
    for (int off = 16; off >= 1; off >>= 1) vs += __shfl_xor(vs, off, 32);
    float rs = rsqrtf(vs * (1.f / DMODEL) + 1e-5f);
#pragma unroll
    for (int i = 0; i < 16; ++i) {
      int d = lane + i * 32, idx = row * DMODEL + d;
      float gamma = sigf(cr[(size_t)row * NCELL + 2560 + d]);
      float y = gamma * ((vals[i] - mean) * rs * ng[d] + nb[d]);
      iout[idx] = y; ioutb[idx] = f2bf(y);
    }
  }
}

/* ---------------- prepack ---------------- */

__global__ void anima_prepack(SRC s, KP p) {
  int job = blockIdx.y;
  int tid = blockIdx.x * blockDim.x + threadIdx.x;
  int str = gridDim.x * blockDim.x;
  switch (job) {
    case 0: for (int i = tid; i < DMODEL * SENS; i += str) ((bf16*)p.wenc)[i] = f2bf(s.wenc_w[i]); break;
    case 1: for (int i = tid; i < DMODEL * 1536; i += str) {
              int o = i / 1536, k = i % 1536;
              float v = (k < 512) ? s.wfw[o * 512 + k]
                      : (k < 1024) ? s.wfi[o * 512 + k - 512] : s.wfa[o * 512 + k - 1024];
              ((bf16*)p.wfrom)[i] = f2bf(v); } break;
    case 2: for (int i = tid; i < DMODEL * 1024; i += str) ((bf16*)p.wgate)[i] = f2bf(s.wgate_w[i]); break;
    case 3: for (int i = tid; i < DMODEL * 1536; i += str) {
              int o = i / 1536, k = i % 1536;
              float v = (k < 512) ? s.ifw[o * 512 + k]
                      : (k < 1024) ? s.ifi[o * 512 + k - 512] : s.ifa[o * 512 + k - 1024];
              ((bf16*)p.ifrom)[i] = f2bf(v); } break;
    case 4: for (int i = tid; i < DMODEL * 1536; i += str) {
              int o = i / 1536, k = i % 1536;
              float v = (k < 512) ? s.afw[o * 512 + k]
                      : (k < 1024) ? s.afi[o * 512 + k - 512] : s.afa[o * 512 + k - 1024];
              ((bf16*)p.afrom)[i] = f2bf(v); } break;
    case 5: { const float* six[6] = { s.fz_w, s.fr_w, s.wa_w, s.wb_w, s.wd_w, s.gam_w };
              for (long i = tid; i < (long)NCELL * 1024; i += str) {
                long r = i >> 10; int k = (int)(i & 1023);
                float v;
                if (r < 3072)       v = six[r >> 9][(r & 511) * 1024 + k];
                else if (r < 11264) v = s.WB_w[(r - 3072) * 1024 + k];
                else                v = s.WC_w[(r - 11264) * 1024 + k];
                ((bf16*)p.cellA)[i] = f2bf(v); } } break;
    case 6: { const float* sb[6] = { s.fz_b, s.fr_b, s.wa_b, s.wb_b, s.wd_b, s.gam_b };
              for (int r = tid; r < NCELL; r += str) {
                float v;
                if (r < 3072)       v = sb[r >> 9][r & 511];
                else if (r < 11264) v = s.WB_b[r - 3072];
                else                v = s.WC_b[r - 11264];
                ((float*)p.cell_bias)[r] = v; } } break;
    case 7:  for (int i = tid; i < DMODEL * 1024; i += str) ((bf16*)p.fasth)[i] = f2bf(s.fh_w[i]); break;
    case 8:  for (int i = tid; i < DMETA * 576; i += str)  ((bf16*)p.metaw)[i] = f2bf(s.meta_w[i]); break;
    case 9:  for (int i = tid; i < DMODEL * 576; i += str) ((bf16*)p.horw)[i] = f2bf(s.hor_w[i]); break;
    case 10: for (int i = tid; i < NBOT * 1024; i += str)  ((bf16*)p.wabs)[i] = f2bf(s.abs_w[i]); break;
    case 11: for (int i = tid; i < DMODEL * NBOT; i += str)((bf16*)p.wexp)[i] = f2bf(s.exp_w[i]); break;
    case 12: for (int i = tid; i < DMODEL * 1024; i += str)((bf16*)p.agate)[i] = f2bf(s.agate_w[i]); break;
    case 13: for (int i = tid; i < NOUT * DMODEL; i += str)((bf16*)p.wphi)[i] = f2bf(s.phi_w[i]); break;
    case 14: for (int i = tid; i < DMODEL * DSTATE; i += str) ((float*)p.negA)[i] = -__expf(s.Alog[i]); break;
    case 15: for (int i = tid; i < T_SEQ * N_BATCH * SENS; i += str) {
               int t = i / (N_BATCH * SENS); int rem = i % (N_BATCH * SENS);
               int b = rem / SENS, k = rem % SENS;
               ((bf16*)p.xT_bf)[i] = f2bf(s.x[((size_t)b * T_SEQ + t) * SENS + k]); } break;
    case 16: { for (int i = tid; i < p.stateF_n; i += str) p.stateF[i] = 0.f;
               for (int i = tid; i < p.stateB_n; i += str) p.stateB[i] = f2bf(0.f);
               if (tid < 2) p.sync[tid] = 0u; } break;
  }
}

/* ---------------- encoder: xenc[t] = tanh(x_t @ Wenc^T + b), all t in parallel ------- */

__global__ void __launch_bounds__(256) anima_encode(KP p) {
  int gw  = (blockIdx.x * blockDim.x + threadIdx.x) >> 5;
  int gws = (gridDim.x * blockDim.x) >> 5;
  int lane = threadIdx.x & 31;
  for (int ti = gw; ti < T_SEQ * 64; ti += gws) {
    int t = ti >> 6, rem = ti & 63, mt = rem & 1, nt = rem >> 1;
    const bf16* A = p.xT_bf + (size_t)t * N_BATCH * SENS;
    f32x8 acc = {};
    for (int kt = 0; kt < SENS / 32; ++kt) {
      bf16x16 a = load_a_frag(A, SENS, mt, kt);
      bf16x16 b = load_b_frag(p.wenc, SENS, nt, kt);
      acc = __builtin_amdgcn_wmma_f32_16x16x32_bf16(false, a, false, b,
                                                    (short)0, acc, false, false);
    }
    int n  = (nt << 4) + (lane & 15);
    int mb = (mt << 4) + ((lane & 16) >> 1);
#pragma unroll
    for (int r = 0; r < 8; ++r) {
      int m = mb + r;
      float v = tanhf(acc[r] + p.wenc_b[n]);
      size_t o = (size_t)t * SZ + (size_t)m * DMODEL + n;
      p.xenc_f[o] = v; p.xenc_bf[o] = f2bf(v);
    }
  }
}

/* ---------------- persistent recurrent scan ---------------- */

__global__ void __launch_bounds__(256) anima_main(KP p) {
  __shared__ bf16 sA[N_BATCH * 1536];   /* 96 KB staging (320 KB LDS/WGP on CDNA5) */
  for (int t = 0; t < T_SEQ; ++t) {
    const int cur = t & 1, nxt = cur ^ 1;
    const bf16 *xe_bf = p.xenc_bf + (size_t)t * SZ;
    const float *xe_f = p.xenc_f + (size_t)t * SZ;
    bf16 *Wb_c = p.Wb + cur * SZ,  *Wb_n = p.Wb + nxt * SZ;
    bf16 *Ifb_c = p.Ifb + cur * SZ, *Ifb_n = p.Ifb + nxt * SZ;
    bf16 *Ab_c = p.Ab + cur * SZ,  *Ab_n = p.Ab + nxt * SZ;

    /* P1: W_coupled = [W,I,A]_state @ Wfrom_cat^T */
    gemm3(sA, Wb_c, 512, Ifb_c, 512, Ab_c, 512, p.wfrom, nullptr, nullptr,
          p.wcoup_f, 512, p.wcoup_bf, 512, 512, 0);
    gsync(p.sync);
    /* P2: Wg = sigmoid([W_enc, W_coupled] @ Wgate^T + b) */
    gemm3(sA, xe_bf, 512, p.wcoup_bf, 512, nullptr, 0, p.wgate, p.wgate_b, nullptr,
          p.wg_f, 512, nullptr, 0, 512, 2);
    gsync(p.sync);
    /* P3: W_new = LN(Wg*W_enc + (1-Wg)*W_state) */
    rowln_gate(p.wg_f, xe_f, p.Wst + cur * SZ, p.wng, p.wnb,
               p.Wst + nxt * SZ, Wb_n);
    gsync(p.sync);
    /* P4: I_input = [W_new, I_fast, A_state] @ Ifrom_cat^T */
    gemm3(sA, Wb_n, 512, Ifb_c, 512, Ab_c, 512, p.ifrom, nullptr, nullptr,
          p.iin_f, 512, p.iin_bf, 512, 512, 0);
    gsync(p.sync);
    /* P5: big cell GEMM: [h_fast, x] -> z,r,alpha,beta,delta,gamma,W_B,W_C (N=19456) */
    gemm3(sA, Ifb_c, 512, p.iin_bf, 512, nullptr, 0, p.cellA, p.cell_bias, nullptr,
          p.cellraw, NCELL, nullptr, 0, NCELL, 0);
    gsync(p.sync);
    /* P6: SSM slow state + C contraction + r*h_fast */
    cell_slow(p.cellraw, p.iin_f, p.Ifst + cur * SZ,
              p.Hslow + (size_t)cur * SZ * DSTATE, p.negA,
              p.Hslow + (size_t)nxt * SZ * DSTATE, p.hso_f, p.rh_bf);
    gsync(p.sync);
    /* P7: h_cand = tanh([r*h_fast, x] @ fast_h^T + b) */
    gemm3(sA, p.rh_bf, 512, p.iin_bf, 512, nullptr, 0, p.fasth, p.fasth_b, nullptr,
          p.hcand_f, 512, nullptr, 0, 512, 1);
    gsync(p.sync);
    /* P8: h_meta_new = tanh([x, h_meta] @ meta^T + b) */
    gemm3(sA, p.iin_bf, 512, p.Hmb + cur * N_BATCH * DMETA, DMETA, nullptr, 0,
          p.metaw, p.meta_b, nullptr,
          p.Hmeta + nxt * N_BATCH * DMETA, DMETA,
          p.Hmb + nxt * N_BATCH * DMETA, DMETA, DMETA, 1);
    gsync(p.sync);
    /* P9: tau = sigmoid([h_meta_new, x] @ hor^T + b) */
    gemm3(sA, p.Hmb + nxt * N_BATCH * DMETA, DMETA, p.iin_bf, 512, nullptr, 0,
          p.horw, p.hor_b, nullptr, p.tau_f, 512, nullptr, 0, 512, 2);
    gsync(p.sync);
    /* P10: fast GRU update + mix + gamma*LN -> I_out */
    cell_mix(p.cellraw, p.hcand_f, p.Ifst + cur * SZ, p.tau_f, p.hso_f, p.iin_f,
             p.Dvec, p.cng, p.cnb,
             p.Ifst + nxt * SZ, Ifb_n, p.iout_f, p.iout_bf);
    gsync(p.sync);
    /* P11: abstr = tanh([W_new, I_out] @ abstract^T + b) */
    gemm3(sA, Wb_n, 512, p.iout_bf, 512, nullptr, 0, p.wabs, p.abs_b, nullptr,
          nullptr, 0, p.abstr_bf, NBOT, NBOT, 1);
    gsync(p.sync);
    /* P12: expanded = tanh(abstr @ expand^T + b) */
    gemm3(sA, p.abstr_bf, NBOT, nullptr, 0, nullptr, 0, p.wexp, p.exp_b, nullptr,
          p.expand_f, 512, nullptr, 0, 512, 1);
    gsync(p.sync);
    /* P13: A_input = [W_new, I_out, A_state] @ Afrom_cat^T + expanded */
    gemm3(sA, Wb_n, 512, p.iout_bf, 512, Ab_c, 512, p.afrom, nullptr, p.expand_f,
          p.ain_f, 512, p.ain_bf, 512, 512, 0);
    gsync(p.sync);
    /* P14: Ag = sigmoid([A_input, A_state] @ A_gate^T + b) */
    gemm3(sA, p.ain_bf, 512, Ab_c, 512, nullptr, 0, p.agate, p.agate_b, nullptr,
          p.ag_f, 512, nullptr, 0, 512, 2);
    gsync(p.sync);
    /* P15: A_new = LN(Ag*A_input + (1-Ag)*A_state) */
    rowln_gate(p.ag_f, p.ain_f, p.Ast + cur * SZ, p.ang, p.anb,
               p.Ast + nxt * SZ, Ab_n);
    gsync(p.sync);
    /* P16: out[:, t, :] = A_new @ phi^T + b */
    gemm3(sA, Ab_n, 512, nullptr, 0, nullptr, 0, p.wphi, p.phi_b, nullptr,
          p.outp + (size_t)t * NOUT, T_SEQ * NOUT, nullptr, 0, NOUT, 0);
    gsync(p.sync);
  }
}

/* ---------------- host launch ---------------- */

extern "C" void kernel_launch(void* const* d_in, const int* in_sizes, int n_in,
                              void* d_out, int out_size, void* d_ws, size_t ws_size,
                              hipStream_t stream) {
  (void)in_sizes; (void)n_in; (void)out_size; (void)ws_size;
  char* ws = (char*)d_ws;
  size_t off = 0;
  auto alloc = [&](size_t bytes) -> char* {
    size_t r = (off + 255) & ~(size_t)255; off = r + bytes; return ws + r;
  };

  SRC s;
  s.x = (const float*)d_in[0];
  s.wenc_w = (const float*)d_in[1];
  const float* wenc_b = (const float*)d_in[2];
  s.wfw = (const float*)d_in[3]; s.wfi = (const float*)d_in[4]; s.wfa = (const float*)d_in[5];
  s.wgate_w = (const float*)d_in[6];
  const float* wgate_b = (const float*)d_in[7];
  const float* wng = (const float*)d_in[8]; const float* wnb = (const float*)d_in[9];
  s.ifw = (const float*)d_in[10]; s.ifi = (const float*)d_in[11]; s.ifa = (const float*)d_in[12];
  s.abs_w = (const float*)d_in[13]; const float* abs_b = (const float*)d_in[14];
  s.exp_w = (const float*)d_in[15]; const float* exp_b = (const float*)d_in[16];
  s.afw = (const float*)d_in[17]; s.afi = (const float*)d_in[18]; s.afa = (const float*)d_in[19];
  s.agate_w = (const float*)d_in[20]; const float* agate_b = (const float*)d_in[21];
  const float* ang = (const float*)d_in[22]; const float* anb = (const float*)d_in[23];
  s.phi_w = (const float*)d_in[24]; const float* phi_b = (const float*)d_in[25];
  s.fz_w = (const float*)d_in[26]; s.fz_b = (const float*)d_in[27];
  s.fr_w = (const float*)d_in[28]; s.fr_b = (const float*)d_in[29];
  s.fh_w = (const float*)d_in[30]; const float* fh_b = (const float*)d_in[31];
  s.wa_w = (const float*)d_in[32]; s.wa_b = (const float*)d_in[33];
  s.wb_w = (const float*)d_in[34]; s.wb_b = (const float*)d_in[35];
  s.wd_w = (const float*)d_in[36]; s.wd_b = (const float*)d_in[37];
  s.Alog = (const float*)d_in[38];
  s.WB_w = (const float*)d_in[39]; s.WB_b = (const float*)d_in[40];
  s.WC_w = (const float*)d_in[41]; s.WC_b = (const float*)d_in[42];
  s.meta_w = (const float*)d_in[43]; const float* meta_b = (const float*)d_in[44];
  s.hor_w = (const float*)d_in[45]; const float* hor_b = (const float*)d_in[46];
  const float* Dvec = (const float*)d_in[47];
  const float* cng = (const float*)d_in[48]; const float* cnb = (const float*)d_in[49];
  s.gam_w = (const float*)d_in[50]; s.gam_b = (const float*)d_in[51];

  KP p;
  p.wenc  = (bf16*)alloc(sizeof(bf16) * DMODEL * SENS);
  p.wfrom = (bf16*)alloc(sizeof(bf16) * DMODEL * 1536);
  p.wgate = (bf16*)alloc(sizeof(bf16) * DMODEL * 1024);
  p.ifrom = (bf16*)alloc(sizeof(bf16) * DMODEL * 1536);
  p.cellA = (bf16*)alloc(sizeof(bf16) * (size_t)NCELL * 1024);
  p.fasth = (bf16*)alloc(sizeof(bf16) * DMODEL * 1024);
  p.metaw = (bf16*)alloc(sizeof(bf16) * DMETA * 576);
  p.horw  = (bf16*)alloc(sizeof(bf16) * DMODEL * 576);
  p.wabs  = (bf16*)alloc(sizeof(bf16) * NBOT * 1024);
  p.wexp  = (bf16*)alloc(sizeof(bf16) * DMODEL * NBOT);
  p.afrom = (bf16*)alloc(sizeof(bf16) * DMODEL * 1536);
  p.agate = (bf16*)alloc(sizeof(bf16) * DMODEL * 1024);
  p.wphi  = (bf16*)alloc(sizeof(bf16) * NOUT * DMODEL);
  p.cell_bias = (float*)alloc(sizeof(float) * NCELL);
  p.negA  = (float*)alloc(sizeof(float) * DMODEL * DSTATE);
  p.xT_bf = (bf16*)alloc(sizeof(bf16) * T_SEQ * N_BATCH * SENS);
  p.xenc_f  = (float*)alloc(sizeof(float) * (size_t)T_SEQ * SZ);
  p.xenc_bf = (bf16*)alloc(sizeof(bf16) * (size_t)T_SEQ * SZ);

  char* stF0 = alloc(sizeof(float) * 2 * SZ);           p.Wst   = (float*)stF0;
  p.Ifst  = (float*)alloc(sizeof(float) * 2 * SZ);
  p.Ast   = (float*)alloc(sizeof(float) * 2 * SZ);
  p.Hmeta = (float*)alloc(sizeof(float) * 2 * N_BATCH * DMETA);
  p.Hslow = (float*)alloc(sizeof(float) * 2 * (size_t)SZ * DSTATE);
  char* stF1 = ws + off;
  p.stateF = (float*)stF0;
  p.stateF_n = (int)(((float*)stF1) - ((float*)stF0));
  char* stB0 = alloc(sizeof(bf16) * 2 * SZ);            p.Wb = (bf16*)stB0;
  p.Ifb = (bf16*)alloc(sizeof(bf16) * 2 * SZ);
  p.Ab  = (bf16*)alloc(sizeof(bf16) * 2 * SZ);
  p.Hmb = (bf16*)alloc(sizeof(bf16) * 2 * N_BATCH * DMETA);
  char* stB1 = ws + off;
  p.stateB = (bf16*)stB0;
  p.stateB_n = (int)(((bf16*)stB1) - ((bf16*)stB0));

  p.wcoup_f = (float*)alloc(sizeof(float) * SZ);
  p.wg_f    = (float*)alloc(sizeof(float) * SZ);
  p.iin_f   = (float*)alloc(sizeof(float) * SZ);
  p.cellraw = (float*)alloc(sizeof(float) * (size_t)N_BATCH * NCELL);
  p.hcand_f = (float*)alloc(sizeof(float) * SZ);
  p.tau_f   = (float*)alloc(sizeof(float) * SZ);
  p.hso_f   = (float*)alloc(sizeof(float) * SZ);
  p.ain_f   = (float*)alloc(sizeof(float) * SZ);
  p.ag_f    = (float*)alloc(sizeof(float) * SZ);
  p.expand_f= (float*)alloc(sizeof(float) * SZ);
  p.iout_f  = (float*)alloc(sizeof(float) * SZ);
  p.wcoup_bf= (bf16*)alloc(sizeof(bf16) * SZ);
  p.iin_bf  = (bf16*)alloc(sizeof(bf16) * SZ);
  p.rh_bf   = (bf16*)alloc(sizeof(bf16) * SZ);
  p.iout_bf = (bf16*)alloc(sizeof(bf16) * SZ);
  p.ain_bf  = (bf16*)alloc(sizeof(bf16) * SZ);
  p.abstr_bf= (bf16*)alloc(sizeof(bf16) * N_BATCH * NBOT);
  p.sync    = (unsigned*)alloc(sizeof(unsigned) * 8);

  p.wenc_b = wenc_b; p.wgate_b = wgate_b; p.wng = wng; p.wnb = wnb;
  p.abs_b = abs_b; p.exp_b = exp_b; p.ang = ang; p.anb = anb;
  p.agate_b = agate_b; p.phi_b = phi_b; p.fasth_b = fh_b;
  p.meta_b = meta_b; p.hor_b = hor_b; p.Dvec = Dvec; p.cng = cng; p.cnb = cnb;
  p.outp = (float*)d_out;

  anima_prepack<<<dim3(512, 17), 256, 0, stream>>>(s, p);
  anima_encode<<<512, 256, 0, stream>>>(p);
  anima_main<<<64, 256, 0, stream>>>(p);
}
